// LambdaLayer_10153302687838
// MI455X (gfx1250) — matrix operand order repaired and checked by reading, not compiled, not run on hardware
//
#include <hip/hip_runtime.h>
#include <hip/hip_bf16.h>

// ---------------------------------------------------------------------------
// LambdaLayer forward for MI455X (gfx1250, wave32, WMMA f32_16x16x32_f16,
// async global->LDS staging with double buffering).
//
//  K0 packw      : pack w_q|w_k|w_v -> Wp f16 [144][256]
//  K1 xt_kernel  : x (B,256,1024) f32 -> xT f16 [32768][256]  (K-contiguous)
//  K2 gather_embT: rel_pos_emb gather -> embT f16 [1024 n][16 k][1024 m]
//  K3 proj       : WMMA GEMM qkv[144][32768] = Wp * xT^T (f32 accum)
//  K4 stats_k    : per-channel mean / inv-std (BN, biased var)
//  K5 softmax_k  : softmax over m for the 16 k-channels (in place)
//  K6 bn_apply   : BN(q) in place; BN(v) -> Vt f16 [2048 cols=(b*64+v)][1024 m]
//  K7 lamc_k     : lam_c[b][16][64] = softmax(k) . v
//  K8 lamp_main  : per n: lam_p(16 x 2048) = embT_n * Vt^T via WMMA,
//                  fuse (+lam_c), contract with q over k, write Y
// ---------------------------------------------------------------------------

typedef __attribute__((ext_vector_type(16))) _Float16 v16h;
typedef __attribute__((ext_vector_type(8)))  _Float16 v8h;
typedef __attribute__((ext_vector_type(8)))  float    v8f;

static __device__ __forceinline__ v16h mk16(v8h a, v8h b) {
  return __builtin_shufflevector(a, b, 0,1,2,3,4,5,6,7,8,9,10,11,12,13,14,15);
}
static __device__ __forceinline__ v8f wmma16(v16h a, v16h b, v8f c) {
  // D = A(16x32 f16) * B(32x16 f16) + C(16x16 f32)
  return __builtin_amdgcn_wmma_f32_16x16x32_f16(false, a, false, b, (short)0, c,
                                                false, false);
}

// ---- CDNA5 async global->LDS copy (ASYNCcnt-tracked) ----------------------
static __device__ __forceinline__ void wait_async0() {
#if __has_builtin(__builtin_amdgcn_s_wait_asynccnt)
  __builtin_amdgcn_s_wait_asynccnt(0);
#else
  asm volatile("s_wait_asynccnt 0x0" ::: "memory");
#endif
}

// Copy 256 contiguous bytes global -> LDS via 16 async b128 ops.
// IOFFSET is added to BOTH the LDS and global addresses (ISA 15.18.3), so a
// single base pair + immediate offsets covers the whole row.
#define ALOAD(off)                                                          \
  asm volatile("global_load_async_to_lds_b128 %0, %1, off offset:" #off     \
               :: "v"(l), "v"(g) : "memory")
static __device__ __forceinline__ void stage256_async(const _Float16* src,
                                                      _Float16* dstLds) {
  // low 32 bits of a generic shared-aperture pointer == LDS byte offset
  unsigned int       l = (unsigned int)(unsigned long long)(uintptr_t)dstLds;
  unsigned long long g = (unsigned long long)(uintptr_t)src;
  ALOAD(0);   ALOAD(16);  ALOAD(32);  ALOAD(48);
  ALOAD(64);  ALOAD(80);  ALOAD(96);  ALOAD(112);
  ALOAD(128); ALOAD(144); ALOAD(160); ALOAD(176);
  ALOAD(192); ALOAD(208); ALOAD(224); ALOAD(240);
}
#undef ALOAD

// ------------------------------ constants ----------------------------------
#define NB     32          // batch
#define NPIX   1024        // 32*32
#define CIN    256
#define NJ     32768       // NB*NPIX
#define LPITCH 136         // f16 pitch of Vt LDS tile (128 + 8 pad)

// ------------------------------ K0: pack W --------------------------------
__global__ void packw(const float* __restrict__ wq, const float* __restrict__ wk,
                      const float* __restrict__ wv, _Float16* __restrict__ Wp) {
  int idx = blockIdx.x * 256 + threadIdx.x;          // 144*256 exact
  int r = idx >> 8, c = idx & 255;
  float v;
  if (r < 64)       v = wq[r * 256 + c];
  else if (r < 80)  v = wk[(r - 64) * 256 + c];
  else              v = wv[(r - 80) * 256 + c];
  Wp[idx] = (_Float16)v;
}

// ------------------------------ K1: x transpose ----------------------------
__global__ __launch_bounds__(256) void xt_kernel(const float* __restrict__ x,
                                                 _Float16* __restrict__ xT) {
  __shared__ float tile[256 * 65];
  const int tid = threadIdx.x;
  const int b  = blockIdx.x >> 4;
  const int p0 = (blockIdx.x & 15) * 64;
  const float* xb = x + (size_t)b * (CIN * NPIX);
  for (int idx = tid; idx < 256 * 64; idx += 256) {
    int c = idx >> 6, p = idx & 63;
    tile[c * 65 + p] = xb[c * NPIX + p0 + p];
  }
  __syncthreads();
  for (int idx = tid; idx < 256 * 64; idx += 256) {
    int p = idx >> 8, c = idx & 255;
    xT[((size_t)b * NPIX + p0 + p) * CIN + c] = (_Float16)tile[c * 65 + p];
  }
}

// ------------------------------ K2: emb gather -----------------------------
__global__ void gather_embT(const float* __restrict__ rpe,
                            _Float16* __restrict__ embT) {
  size_t idx = (size_t)blockIdx.x * 256 + threadIdx.x;   // 1024*16*1024 exact
  int m = (int)(idx & 1023);
  int k = (int)((idx >> 10) & 15);
  int n = (int)(idx >> 14);
  int rn = n >> 5, cn = n & 31, rm = m >> 5, cm = m & 31;
  int ri = rm - rn + 31, ci = cm - cn + 31;              // [0, 62]
  embT[idx] = (_Float16)rpe[((size_t)ri * 63 + ci) * 16 + k];
}

// ------------------------------ K3: projection GEMM ------------------------
// qkv[144][32768] = Wp(144x256) * xT(32768x256)^T, 9 waves, one M-tile each.
__global__ __launch_bounds__(288) void proj(const _Float16* __restrict__ Wp,
                                            const _Float16* __restrict__ xT,
                                            float* __restrict__ qkv) {
  __shared__ __align__(16) _Float16 xl[128 * 264];       // 67.6 KB
  const int tid  = threadIdx.x;
  const int wave = tid >> 5;
  const int lane = tid & 31;
  const int lmod = lane & 15, lhalf = lane >> 4;
  const int j0 = blockIdx.x * 128;

  if (tid < 256) {                                       // stage xT tile (async)
    int row = tid >> 1, half = (tid & 1) * 128;
    stage256_async(xT + (size_t)(j0 + row) * CIN + half, xl + row * 264 + half);
  }
  wait_async0();
  __syncthreads();

  const int r0 = wave * 16;                              // M-tile rows
  v8f zero = {};
  v8f acc[8];
#pragma unroll
  for (int t = 0; t < 8; ++t) acc[t] = zero;

#pragma unroll
  for (int ks = 0; ks < 256; ks += 32) {
    const _Float16* ap = Wp + (size_t)(r0 + lmod) * CIN + ks + lhalf * 16;
    v16h a = mk16(*(const v8h*)ap, *(const v8h*)(ap + 8));
#pragma unroll
    for (int t = 0; t < 8; ++t) {
      const _Float16* bp = xl + (t * 16 + lmod) * 264 + ks + lhalf * 16;
      v16h b = mk16(*(const v8h*)bp, *(const v8h*)(bp + 8));
      acc[t] = wmma16(a, b, acc[t]);
    }
  }
  // C layout: VGPR r -> M = r (+8 for upper lane half), N = lane%16
#pragma unroll
  for (int t = 0; t < 8; ++t) {
    int j = j0 + t * 16 + lmod;
#pragma unroll
    for (int r = 0; r < 8; ++r) {
      int ch = r0 + r + lhalf * 8;
      qkv[(size_t)ch * NJ + j] = acc[t][r];
    }
  }
}

// ------------------------------ K4: BN stats -------------------------------
__global__ __launch_bounds__(256) void stats_k(const float* __restrict__ qkv,
                                               float* __restrict__ st) {
  const int ch = blockIdx.x;
  const int tid = threadIdx.x;
  const float* row = qkv + (size_t)ch * NJ;
  float s = 0.f, s2 = 0.f;
  for (int i = tid; i < NJ; i += 256) { float v = row[i]; s += v; s2 += v * v; }
  __shared__ float rs[256], rq[256];
  rs[tid] = s; rq[tid] = s2; __syncthreads();
  for (int off = 128; off > 0; off >>= 1) {
    if (tid < off) { rs[tid] += rs[tid + off]; rq[tid] += rq[tid + off]; }
    __syncthreads();
  }
  if (tid == 0) {
    float mean = rs[0] * (1.f / NJ);
    float var  = rq[0] * (1.f / NJ) - mean * mean;       // biased var
    st[ch * 2 + 0] = mean;
    st[ch * 2 + 1] = rsqrtf(var + 1e-5f);
  }
}

// ------------------------------ K5: softmax over m -------------------------
__global__ __launch_bounds__(256) void softmax_k(float* __restrict__ qkv) {
  const int tid = threadIdx.x;
  const int b = blockIdx.x >> 4, kc = blockIdx.x & 15;
  float* row = qkv + (size_t)(64 + kc) * NJ + b * NPIX;
  __shared__ float red[256];
  float m = -3.4e38f;
  for (int i = tid; i < NPIX; i += 256) m = fmaxf(m, row[i]);
  red[tid] = m; __syncthreads();
  for (int off = 128; off > 0; off >>= 1) {
    if (tid < off) red[tid] = fmaxf(red[tid], red[tid + off]);
    __syncthreads();
  }
  float mx = red[0];
  __syncthreads();
  float s = 0.f;
  for (int i = tid; i < NPIX; i += 256) {
    float e = __expf(row[i] - mx);
    row[i] = e; s += e;
  }
  red[tid] = s; __syncthreads();
  for (int off = 128; off > 0; off >>= 1) {
    if (tid < off) red[tid] += red[tid + off];
    __syncthreads();
  }
  float inv = 1.f / red[0];
  for (int i = tid; i < NPIX; i += 256) row[i] *= inv;
}

// ------------------------------ K6: BN apply -------------------------------
__global__ void bn_apply(float* __restrict__ qkv, const float* __restrict__ st,
                         const float* __restrict__ gq, const float* __restrict__ bq,
                         const float* __restrict__ gv, const float* __restrict__ bv,
                         _Float16* __restrict__ Vt) {
  size_t idx = (size_t)blockIdx.x * 256 + threadIdx.x;   // 128*32768 exact
  int ch = (int)(idx >> 15);
  int j  = (int)(idx & 32767);
  if (ch < 64) {                                          // q in place
    float mean = st[ch * 2], is = st[ch * 2 + 1];
    float* p = qkv + (size_t)ch * NJ + j;
    *p = (*p - mean) * is * gq[ch] + bq[ch];
  } else {                                                // v -> Vt f16
    int vc = ch - 64, c2 = 80 + vc;
    float mean = st[c2 * 2], is = st[c2 * 2 + 1];
    float raw = qkv[(size_t)c2 * NJ + j];
    float bn = (raw - mean) * is * gv[vc] + bv[vc];
    int b = j >> 10, p = j & 1023;
    Vt[((size_t)(b * 64 + vc)) * NPIX + p] = (_Float16)bn;
  }
}

// ------------------------------ K7: content lambda -------------------------
__global__ __launch_bounds__(256) void lamc_k(const float* __restrict__ qkv,
                                              const _Float16* __restrict__ Vt,
                                              float* __restrict__ lamc) {
  const int b = blockIdx.x;
#pragma unroll
  for (int pp = 0; pp < 4; ++pp) {
    int id = pp * 256 + threadIdx.x;                      // 1024 (k,v) pairs
    int k = id >> 6, v = id & 63;
    const float*    kr = qkv + (size_t)(64 + k) * NJ + b * NPIX;
    const _Float16* vr = Vt + (size_t)(b * 64 + v) * NPIX;
    float s = 0.f;
    for (int m = 0; m < NPIX; ++m) s += kr[m] * (float)vr[m];
    lamc[((size_t)b * 16 + k) * 64 + v] = s;
  }
}

// ------------------------------ K8: main WMMA kernel -----------------------
// Grid (128, 8): blockIdx.x -> n-group of 8 (one n per wave),
//                blockIdx.y -> 256-column tile of Vt (4 batches * 64 v).
// Each wave: lam_p tile (16 k x 256 cols) as 16 WMMA C-tiles (128 acc VGPRs).
// K = m = 1024 in 8 chunks of 128, double-buffered LDS filled by async DMA.
__global__ __launch_bounds__(256) void lamp_main(const _Float16* __restrict__ embT,
                                                 const _Float16* __restrict__ Vt,
                                                 const float* __restrict__ qn,
                                                 const float* __restrict__ lamc,
                                                 float* __restrict__ out) {
  __shared__ __align__(16) unsigned char smem[2 * 256 * LPITCH * 2];  // 139264 B
  _Float16* buf0 = (_Float16*)smem;
  _Float16* buf1 = buf0 + 256 * LPITCH;
  float*    yws  = (float*)smem;                         // reused in epilogue

  const int tid   = threadIdx.x;
  const int wave  = tid >> 5;
  const int lane  = tid & 31;
  const int lmod  = lane & 15, lhalf = lane >> 4;
  const int n0    = blockIdx.x * 8;
  const int myn   = n0 + wave;
  const int colB  = blockIdx.y * 256;

  v8f zero = {};
  v8f acc[16];
#pragma unroll
  for (int t = 0; t < 16; ++t) acc[t] = zero;

  const _Float16* Arow  = embT + (size_t)myn * (16 * NPIX);
  const _Float16* myVt  = Vt + (size_t)(colB + tid) * NPIX;   // this thread's row

  // prologue: DMA chunk 0 into buf0
  stage256_async(myVt, buf0 + tid * LPITCH);
  wait_async0();
  __syncthreads();

  for (int mc = 0; mc < NPIX; mc += 128) {
    _Float16* cur = (mc & 128) ? buf1 : buf0;
    _Float16* nxt = (mc & 128) ? buf0 : buf1;
    if (mc + 128 < NPIX)                                  // DMA next chunk
      stage256_async(myVt + mc + 128, nxt + tid * LPITCH);

#pragma unroll
    for (int ks = 0; ks < 128; ks += 32) {
      const _Float16* ap = Arow + (size_t)lmod * NPIX + (mc + ks) + lhalf * 16;
      v16h a = mk16(*(const v8h*)ap, *(const v8h*)(ap + 8));
#pragma unroll
      for (int t = 0; t < 16; ++t) {
        const _Float16* bp = cur + (t * 16 + lmod) * LPITCH + ks + lhalf * 16;
        v16h b = mk16(*(const v8h*)bp, *(const v8h*)(bp + 8));
        acc[t] = wmma16(a, b, acc[t]);
      }
    }
    wait_async0();          // my DMA for 'nxt' done
    __syncthreads();        // everyone done computing 'cur' + loading 'nxt'
  }

  // Epilogue: lam_tot = lam_p + lam_c ; Y[b][h*64+v][n] = sum_k q[b][h][k][n]*lam_tot
  const int koff = lhalf * 8;      // C tile: VGPR r -> k = r + koff, N = lane%16
#pragma unroll
  for (int t = 0; t < 16; ++t) {
    int colg = colB + t * 16 + lmod;
    int b = colg >> 6, vv = colg & 63;
    float a[8];
#pragma unroll
    for (int r = 0; r < 8; ++r)
      a[r] = acc[t][r] + lamc[((size_t)b * 16 + (r + koff)) * 64 + vv];
#pragma unroll
    for (int h = 0; h < 4; ++h) {
      float s = 0.f;
#pragma unroll
      for (int r = 0; r < 8; ++r)
        s += qn[(size_t)(h * 16 + r + koff) * NJ + b * NPIX + myn] * a[r];
      s += __shfl_xor(s, 16);                       // combine k-halves
      if (lhalf == 0) yws[(t * 16 + lmod) * 32 + h * 8 + wave] = s;
    }
  }
  __syncthreads();
  {   // store: thread = one column, 4 heads, 8 contiguous n (32 B runs)
    int colg = colB + tid;
    int b = colg >> 6, vv = colg & 63;
#pragma unroll
    for (int h = 0; h < 4; ++h) {
      float4 lo = *(float4*)(yws + tid * 32 + h * 8);
      float4 hi = *(float4*)(yws + tid * 32 + h * 8 + 4);
      float* op = out + ((size_t)b * 256 + h * 64 + vv) * NPIX + n0;
      *(float4*)op = lo;
      *(float4*)(op + 4) = hi;
    }
  }
}

// ------------------------------ launcher -----------------------------------
extern "C" void kernel_launch(void* const* d_in, const int* in_sizes, int n_in,
                              void* d_out, int out_size, void* d_ws, size_t ws_size,
                              hipStream_t stream) {
  const float* x   = (const float*)d_in[0];
  const float* w_q = (const float*)d_in[1];
  const float* w_k = (const float*)d_in[2];
  const float* w_v = (const float*)d_in[3];
  const float* gq  = (const float*)d_in[4];
  const float* bq  = (const float*)d_in[5];
  const float* gv  = (const float*)d_in[6];
  const float* bv  = (const float*)d_in[7];
  const float* rpe = (const float*)d_in[8];
  float* out = (float*)d_out;

  char* ws = (char*)d_ws;
  _Float16* Wp   = (_Float16*)(ws + 0);                  //    73,728 B
  _Float16* embT = (_Float16*)(ws + 73728);              // 33,554,432 B
  _Float16* xT   = (_Float16*)(ws + 33628160);           // 16,777,216 B
  float*    qkv  = (float*)   (ws + 50405376);           // 18,874,368 B
  float*    st   = (float*)   (ws + 69279744);           //     1,280 B
  _Float16* Vt   = (_Float16*)(ws + 69281024);           //  4,194,304 B
  float*    lamc = (float*)   (ws + 73475328);           //    131,072 B
                                                         // total ~70.2 MB

  packw      <<<144,   256, 0, stream>>>(w_q, w_k, w_v, Wp);
  xt_kernel  <<<512,   256, 0, stream>>>(x, xT);
  gather_embT<<<65536, 256, 0, stream>>>(rpe, embT);
  proj       <<<256,   288, 0, stream>>>(Wp, xT, qkv);
  stats_k    <<<144,   256, 0, stream>>>(qkv, st);
  softmax_k  <<<512,   256, 0, stream>>>(qkv);
  bn_apply   <<<16384, 256, 0, stream>>>(qkv, st, gq, bq, gv, bv, Vt);
  lamc_k     <<<32,    256, 0, stream>>>(qkv, Vt, lamc);
  dim3 g(128, 8);
  lamp_main  <<<g,     256, 0, stream>>>(embT, Vt, qkv, lamc, out);
}
